// FLB_Attention_Layer_21311627722792
// MI455X (gfx1250) — matrix operand order repaired and checked by reading, not compiled, not run on hardware
//
#include <hip/hip_runtime.h>
#include <hip/hip_bf16.h>

// ---------------------------------------------------------------------------
// FLB Attention Layer for MI455X (gfx1250)
//   bf16 WMMA GEMMs, double-buffered async global->LDS staging, fused epilogues
//   B=4096, D=2048, H=16, DH=128
// ---------------------------------------------------------------------------

typedef __attribute__((ext_vector_type(16))) __bf16 v16bf;
typedef __attribute__((ext_vector_type(8)))  float  v8f;

#define BDIM 4096
#define DDIM 2048
#define HNUM 16
#define DHD  128
#define KDIM 2048
#define NDIM 2048
#define LDS_STRIDE 40   // 32 K-elems + 8 pad (bf16 units) -> 80B rows, 16B aligned
#define LDS_BUF_BYTES (128 * LDS_STRIDE * 2)   // 10240 B per matrix buffer

struct U32B { uint4 lo, hi; };

static __device__ __forceinline__ v16bf frag_cast(uint4 lo, uint4 hi) {
  U32B t{lo, hi};
  return __builtin_bit_cast(v16bf, t);
}

// f32 -> bf16, round-to-nearest-even
static __device__ __forceinline__ unsigned short f2bf(float f) {
  unsigned u = __builtin_bit_cast(unsigned, f);
  if ((u & 0x7fffffffu) > 0x7f800000u) return (unsigned short)((u >> 16) | 0x40u);
  u += 0x7fffu + ((u >> 16) & 1u);
  return (unsigned short)(u >> 16);
}

static __device__ __forceinline__ unsigned pack_bf2(float a, float b) {
  return (unsigned)f2bf(a) | ((unsigned)f2bf(b) << 16);
}

// fast transcendentals (gfx1250 has TANH/RCP trans ops)
#if __has_builtin(__builtin_amdgcn_tanhf)
static __device__ __forceinline__ float fast_tanh(float x) { return __builtin_amdgcn_tanhf(x); }
#else
static __device__ __forceinline__ float fast_tanh(float x) { return tanhf(x); }
#endif
static __device__ __forceinline__ float fast_rcp(float x) { return __builtin_amdgcn_rcpf(x); }

// raw LDS byte offset of a pointer into __shared__ (for async-to-LDS asm)
static __device__ __forceinline__ unsigned lds_off(const void* p) {
  return (unsigned)(unsigned long long)(const __attribute__((address_space(3))) char*)p;
}

template <int N>
static __device__ __forceinline__ void wait_asynccnt() {
  asm volatile("s_wait_asynccnt %0" ::"n"(N));
}

// one lane: copy 2 x 16B from global to LDS via the async DMA path (ASYNCcnt)
static __device__ __forceinline__ void async_copy32(unsigned ldsOff,
                                                    unsigned long long gaddr) {
  asm volatile(
      "global_load_async_to_lds_b128 %0, %1, off\n\t"
      "global_load_async_to_lds_b128 %0, %1, off offset:16"
      ::"v"(ldsOff), "v"(gaddr)
      : "memory");
}

// ---------------------------------------------------------------------------
// cvt kernels (vectorized: 4 f32 -> 4 bf16 per thread)
// ---------------------------------------------------------------------------
__global__ void cvt_f32_bf16_v4(const float4* __restrict__ src,
                                uint2* __restrict__ dst, int n4) {
  int i = blockIdx.x * 256 + threadIdx.x;
  if (i < n4) {
    float4 f = src[i];
    uint2 o;
    o.x = pack_bf2(f.x, f.y);
    o.y = pack_bf2(f.z, f.w);
    dst[i] = o;
  }
}

// write fdbk rows into packed tokens[B,3,D] at t=2
__global__ void cvt_fdbk_v4(const float4* __restrict__ src,
                            unsigned short* __restrict__ tokens) {
  int i = blockIdx.x * 256 + threadIdx.x;  // exact grid: B*D/4 threads
  int b  = i >> 9;                         // / (DDIM/4)
  int d4 = i & 511;
  float4 f = src[i];
  uint2 o;
  o.x = pack_bf2(f.x, f.y);
  o.y = pack_bf2(f.z, f.w);
  *(uint2*)(tokens + ((size_t)(3 * b + 2)) * DDIM + d4 * 4) = o;
}

// ---------------------------------------------------------------------------
// Tiled bf16 GEMM:  Out[M,N] = A[M,K] * W[N,K]^T   (torch Linear convention)
// Block: 256 thr = 8 waves; tile 128x128, K-step 32; wave = 64x32 patch.
// Staging: double-buffered LDS filled with GLOBAL_LOAD_ASYNC_TO_LDS_B128,
//          next tile's DMA overlaps current tile's 8 WMMAs.
// MODE 0: obf[row*ldobf+col] = bf16( sigmoid(acc + bias[col]) * mul[row*D+col] )
// MODE 1: ofp[row*N+col]     = acc
// MODE 2: dout[t*B*D + b*D + col] = acc + res_t[b*D+col],  row = 3b+t
// ---------------------------------------------------------------------------
template <int MODE>
__global__ __launch_bounds__(256)
void gemm_bf16(const unsigned short* __restrict__ A, int lda,
               const unsigned short* __restrict__ Wt,
               const float* __restrict__ bias,
               const float* __restrict__ mul,
               unsigned short* __restrict__ obf, int ldobf,
               float* __restrict__ ofp,
               const float* __restrict__ rx, const float* __restrict__ rl,
               const float* __restrict__ rf, float* __restrict__ dout) {
  __shared__ unsigned short As[2][128 * LDS_STRIDE];
  __shared__ unsigned short Bs[2][128 * LDS_STRIDE];

  const int colBase = blockIdx.x * 128;
  const int rowBase = blockIdx.y * 128;
  const int tid   = threadIdx.x;
  const int lane  = tid & 31;
  const int wave  = tid >> 5;
  const int waveM = wave & 1;   // 2 waves along M (64 rows each)
  const int waveN = wave >> 1;  // 4 waves along N (32 cols each)

  // staging assignment: thread t owns 16+16 bf16 of one tile row
  const int ldr = tid >> 1;     // 0..127
  const int ldh = tid & 1;      // 0..1
  const unsigned long long aSrc =
      (unsigned long long)(A + (size_t)(rowBase + ldr) * lda + ldh * 16);
  const unsigned long long bSrc =
      (unsigned long long)(Wt + (size_t)(colBase + ldr) * KDIM + ldh * 16);
  const unsigned aOff = lds_off(&As[0][ldr * LDS_STRIDE + ldh * 16]);
  const unsigned bOff = lds_off(&Bs[0][ldr * LDS_STRIDE + ldh * 16]);

  // WMMA fragment addressing (16-bit A 16x32 / B 32x16 layouts)
  const int lrow = lane & 15;
  const int kbA  = (lane >> 4) * 16;  // A: 16 contiguous k at kbA
  const int ksB  = (lane >> 4) * 16;  // B: 16 contiguous k at ksB

  v8f acc[4][2];
#pragma unroll
  for (int i = 0; i < 4; i++)
#pragma unroll
    for (int j = 0; j < 2; j++)
      acc[i][j] = v8f{0.f, 0.f, 0.f, 0.f, 0.f, 0.f, 0.f, 0.f};

  constexpr int NT = KDIM / 32;

  // prime buffer 0 with tile 0 (4 async b128 per thread => ASYNCcnt=4/wave)
  async_copy32(aOff, aSrc);
  async_copy32(bOff, bSrc);

  for (int kt = 0; kt < NT; ++kt) {
    __syncthreads();  // all waves done reading the buffer we are about to fill
    if (kt + 1 < NT) {
      const unsigned bufsel = ((kt + 1) & 1) * LDS_BUF_BYTES;
      const unsigned long long koff = (unsigned long long)(kt + 1) * 64;  // 32 bf16
      async_copy32(aOff + bufsel, aSrc + koff);
      async_copy32(bOff + bufsel, bSrc + koff);
      wait_asynccnt<4>();  // in-order: tile kt has fully landed
    } else {
      wait_asynccnt<0>();
    }
    __syncthreads();  // tile kt visible to every wave

    const int buf = kt & 1;
    v16bf afrag[4], bfrag[2];
#pragma unroll
    for (int tm = 0; tm < 4; tm++) {
      const uint4* p =
          (const uint4*)&As[buf][(waveM * 64 + tm * 16 + lrow) * LDS_STRIDE + kbA];
      afrag[tm] = frag_cast(p[0], p[1]);  // k=kbA..+15 contiguous
    }
#pragma unroll
    for (int tn = 0; tn < 2; tn++) {
      const uint4* p =
          (const uint4*)&Bs[buf][(waveN * 32 + tn * 16 + lrow) * LDS_STRIDE + ksB];
      bfrag[tn] = frag_cast(p[0], p[1]);  // k=ksB..+15 contiguous
    }
#pragma unroll
    for (int tm = 0; tm < 4; tm++)
#pragma unroll
      for (int tn = 0; tn < 2; tn++)
        acc[tm][tn] = __builtin_amdgcn_wmma_f32_16x16x32_bf16(
            false, afrag[tm], false, bfrag[tn], (short)0, acc[tm][tn],
            false, false);
  }

  // epilogue: C 16x16 f32 layout -> m = r + (lane>>4)*8, n = lane&15
  const int wRow = rowBase + waveM * 64;
  const int wCol = colBase + waveN * 32;
  const int nloc = lane & 15;
  const int moff = (lane >> 4) * 8;
#pragma unroll
  for (int tm = 0; tm < 4; tm++) {
#pragma unroll
    for (int tn = 0; tn < 2; tn++) {
      const int ncol = wCol + tn * 16 + nloc;
#pragma unroll
      for (int r = 0; r < 8; r++) {
        const int mrow = wRow + tm * 16 + moff + r;
        float v = acc[tm][tn][r];
        if (MODE == 0) {
          float g = fast_rcp(1.0f + __expf(-(v + bias[ncol])));
          float o = g * mul[(size_t)mrow * DDIM + ncol];
          obf[(size_t)mrow * ldobf + ncol] = f2bf(o);
        } else if (MODE == 1) {
          ofp[(size_t)mrow * NDIM + ncol] = v;
        } else {
          int b = mrow / 3;
          int t = mrow - 3 * b;
          const float* res = (t == 0) ? rx : (t == 1) ? rl : rf;
          dout[(size_t)t * (size_t)BDIM * DDIM + (size_t)b * DDIM + ncol] =
              v + res[(size_t)b * DDIM + ncol];
        }
      }
    }
  }
}

// ---------------------------------------------------------------------------
// Additive (Bahdanau) attention: one wave32 per (b,h); dh=128 -> 4 f32/lane.
// score[i][j] = v_a[h] . tanh(q_i + k_j); softmax over j; out = w @ v (bf16).
// ---------------------------------------------------------------------------
__global__ __launch_bounds__(256)
void attn_kernel(const float* __restrict__ qf, const float* __restrict__ kf,
                 const float* __restrict__ vf, const float* __restrict__ va,
                 unsigned short* __restrict__ abf) {
  const int lane = threadIdx.x & 31;
  const int wv   = threadIdx.x >> 5;
  const int gid  = blockIdx.x * 8 + wv;  // [0, B*H)
  const int b = gid >> 4;
  const int h = gid & (HNUM - 1);
  const size_t hb = (size_t)h * DHD + lane * 4;

  float4 vA = *(const float4*)(va + hb);
  float4 Q[3], Kv[3], Vv[3];
#pragma unroll
  for (int i = 0; i < 3; i++) {
    size_t base = ((size_t)(b * 3 + i)) * DDIM + hb;
    Q[i]  = *(const float4*)(qf + base);
    Kv[i] = *(const float4*)(kf + base);
    Vv[i] = *(const float4*)(vf + base);
  }

  float s[3][3];
#pragma unroll
  for (int i = 0; i < 3; i++)
#pragma unroll
    for (int j = 0; j < 3; j++) {
      float p = vA.x * fast_tanh(Q[i].x + Kv[j].x)
              + vA.y * fast_tanh(Q[i].y + Kv[j].y)
              + vA.z * fast_tanh(Q[i].z + Kv[j].z)
              + vA.w * fast_tanh(Q[i].w + Kv[j].w);
      p += __shfl_xor(p, 16, 32);
      p += __shfl_xor(p, 8, 32);
      p += __shfl_xor(p, 4, 32);
      p += __shfl_xor(p, 2, 32);
      p += __shfl_xor(p, 1, 32);
      s[i][j] = p;
    }

#pragma unroll
  for (int i = 0; i < 3; i++) {
    float m  = fmaxf(s[i][0], fmaxf(s[i][1], s[i][2]));
    float e0 = __expf(s[i][0] - m);
    float e1 = __expf(s[i][1] - m);
    float e2 = __expf(s[i][2] - m);
    float inv = fast_rcp(e0 + e1 + e2);
    float w0 = e0 * inv, w1 = e1 * inv, w2 = e2 * inv;
    float ox = w0 * Vv[0].x + w1 * Vv[1].x + w2 * Vv[2].x;
    float oy = w0 * Vv[0].y + w1 * Vv[1].y + w2 * Vv[2].y;
    float oz = w0 * Vv[0].z + w1 * Vv[1].z + w2 * Vv[2].z;
    float ow = w0 * Vv[0].w + w1 * Vv[1].w + w2 * Vv[2].w;
    size_t base = ((size_t)(b * 3 + i)) * DDIM + hb;
    uint2 pk;
    pk.x = pack_bf2(ox, oy);
    pk.y = pack_bf2(oz, ow);
    *(uint2*)(abf + base) = pk;
  }
}

// ---------------------------------------------------------------------------
// host-side orchestration
// ---------------------------------------------------------------------------
extern "C" void kernel_launch(void* const* d_in, const int* in_sizes, int n_in,
                              void* d_out, int out_size, void* d_ws,
                              size_t ws_size, hipStream_t stream) {
  (void)in_sizes; (void)n_in; (void)out_size; (void)ws_size;
  const float* x    = (const float*)d_in[0];
  const float* lat  = (const float*)d_in[1];
  const float* fdbk = (const float*)d_in[2];
  const float* WgL  = (const float*)d_in[3];
  const float* bgL  = (const float*)d_in[4];
  const float* WgX  = (const float*)d_in[5];
  const float* bgX  = (const float*)d_in[6];
  const float* Wq   = (const float*)d_in[7];
  const float* Wk   = (const float*)d_in[8];
  const float* Wv   = (const float*)d_in[9];
  const float* Wo   = (const float*)d_in[10];
  const float* va   = (const float*)d_in[11];
  float* out = (float*)d_out;

  const size_t SZW = (size_t)DDIM * DDIM;      // 4M elems per weight
  const size_t TOK = (size_t)3 * BDIM * DDIM;  // 25.2M elems

  char* p = (char*)d_ws;
  size_t off = 0;
  auto take = [&](size_t bytes) -> void* {
    void* r = p + off;
    off = (off + bytes + 255) & ~(size_t)255;
    return r;
  };
  unsigned short* wbf[6];
  for (int i = 0; i < 6; i++) wbf[i] = (unsigned short*)take(SZW * 2);
  unsigned short* tokens = (unsigned short*)take(TOK * 2);  // [B,3,D] bf16
  float* qf = (float*)take(TOK * 4);
  float* kf = (float*)take(TOK * 4);
  float* vf = (float*)take(TOK * 4);
  unsigned short* abf = (unsigned short*)take(TOK * 2);

  // 1) weights f32 -> bf16 (vectorized x4)
  const float* wsrc[6] = {WgL, WgX, Wq, Wk, Wv, Wo};
  int cvtBlocks = (int)((SZW / 4 + 255) / 256);
  for (int i = 0; i < 6; i++)
    cvt_f32_bf16_v4<<<cvtBlocks, 256, 0, stream>>>(
        (const float4*)wsrc[i], (uint2*)wbf[i], (int)(SZW / 4));

  // 2) fdbk token -> tokens[:,2,:]
  cvt_fdbk_v4<<<(BDIM * DDIM / 4) / 256, 256, 0, stream>>>(
      (const float4*)fdbk, tokens);

  // 3) gate GEMMs (fused sigmoid * token epilogue, bf16 out into tokens)
  dim3 gGate(NDIM / 128, BDIM / 128);  // 16 x 32
  gemm_bf16<0><<<gGate, 256, 0, stream>>>(
      tokens + 2 * DDIM, 3 * DDIM, wbf[0], bgL, lat,
      tokens + DDIM, 3 * DDIM, nullptr, nullptr, nullptr, nullptr, nullptr);
  gemm_bf16<0><<<gGate, 256, 0, stream>>>(
      tokens + DDIM, 3 * DDIM, wbf[1], bgX, x,
      tokens, 3 * DDIM, nullptr, nullptr, nullptr, nullptr, nullptr);

  // 4) Q/K/V projections (f32 out)
  dim3 gQKV(NDIM / 128, (3 * BDIM) / 128);  // 16 x 96
  gemm_bf16<1><<<gQKV, 256, 0, stream>>>(
      tokens, DDIM, wbf[2], nullptr, nullptr, nullptr, 0, qf,
      nullptr, nullptr, nullptr, nullptr);
  gemm_bf16<1><<<gQKV, 256, 0, stream>>>(
      tokens, DDIM, wbf[3], nullptr, nullptr, nullptr, 0, kf,
      nullptr, nullptr, nullptr, nullptr);
  gemm_bf16<1><<<gQKV, 256, 0, stream>>>(
      tokens, DDIM, wbf[4], nullptr, nullptr, nullptr, 0, vf,
      nullptr, nullptr, nullptr, nullptr);

  // 5) additive attention (one wave per (b,h))
  attn_kernel<<<(BDIM * HNUM) / 8, 256, 0, stream>>>(qf, kf, vf, va, abf);

  // 6) output projection + residual scatter into the 3 output planes
  gemm_bf16<2><<<gQKV, 256, 0, stream>>>(
      abf, DDIM, wbf[5], nullptr, nullptr, nullptr, 0, nullptr,
      x, lat, fdbk, out);
}
